// MPNN_Encoder_26276609917010
// MI455X (gfx1250) — matrix operand order repaired and checked by reading, not compiled, not run on hardware
//
#include <hip/hip_runtime.h>
#include <hip/hip_bf16.h>

typedef __attribute__((ext_vector_type(16))) _Float16 v16h;
typedef __attribute__((ext_vector_type(8)))  float    v8f;

#define GN      100000
#define GE      1600000
#define NFEAT   200
#define NHID    128
#define NOUT    64
#define BN_EPS  1e-5f

// ---------------------------------------------------------------- utilities

__global__ __launch_bounds__(256) void k_fill(float* p, float v, int n) {
    int t = blockIdx.x * 256 + threadIdx.x;
    if (t < n) p[t] = v;
}

__global__ __launch_bounds__(256) void k_zero4(float4* p, int n4) {
    int t = blockIdx.x * 256 + threadIdx.x;
    if (t < n4) p[t] = make_float4(0.f, 0.f, 0.f, 0.f);
}

// weighted in-degree accumulation (self loop contributes the initial 1.0)
__global__ __launch_bounds__(256) void k_deg(const int* __restrict__ dst,
                                             const float* __restrict__ w,
                                             float* deg, int e) {
    int t = blockIdx.x * 256 + threadIdx.x;
    if (t < e) atomicAdd(&deg[dst[t]], w[t]);
}

__global__ __launch_bounds__(256) void k_rsqrt(float* d, int n) {
    int t = blockIdx.x * 256 + threadIdx.x;
    if (t < n) { float v = d[t]; d[t] = (v > 0.f) ? rsqrtf(v) : 0.f; }
}

// ---------------------------------------------------------------- B prepack
// Pack row-major f32 W[K][Nout] into per-lane WMMA B fragments (f16):
// layout [kChunk][nTile][lane(32)][16 halves], halves j -> K = kc*32+(lane/16)*8+(j&7)+(j>>3)*16,
// N = nTile*16 + lane%16.  One contiguous 32B load per lane per chunk in the GEMM.
__global__ __launch_bounds__(256) void k_packB(const float* __restrict__ W,
                                               _Float16* __restrict__ Bp,
                                               int K, int Nout) {
    int nt = Nout >> 4;
    int kchunks = (K + 31) >> 5;
    int total = kchunks * nt * 32 * 16;
    int t = blockIdx.x * 256 + threadIdx.x;
    if (t >= total) return;
    int j    = t & 15;
    int lane = (t >> 4) & 31;
    int ctk  = t >> 9;
    int ct   = ctk % nt;
    int kc   = ctk / nt;
    int k = kc * 32 + ((lane >> 4) << 3) + (j & 7) + ((j >> 3) << 4);
    int n = ct * 16 + (lane & 15);
    Bp[t] = (k < K) ? (_Float16)W[(size_t)k * Nout + n] : (_Float16)0.0f;
}

// ---------------------------------------------------------------- WMMA GEMM
// A fragment (16x32 f16, ISA 7.12.2): lane row m = lane%16,
// halves 0..7 = K kb..kb+7, halves 8..15 = K kb+16..kb+23, kb = kc*32 + (lane/16)*8.

// unguarded: 8 consecutive f32 -> f16 halves (two b128 loads)
__device__ __forceinline__ void cvt8(v16h& d, int off, const float* __restrict__ p) {
    const float4 a = *(const float4*)p;
    const float4 b = *(const float4*)(p + 4);
    d[off + 0] = (_Float16)a.x; d[off + 1] = (_Float16)a.y;
    d[off + 2] = (_Float16)a.z; d[off + 3] = (_Float16)a.w;
    d[off + 4] = (_Float16)b.x; d[off + 5] = (_Float16)b.y;
    d[off + 6] = (_Float16)b.z; d[off + 7] = (_Float16)b.w;
}

template<int K, int NT, bool RELU, bool HASBIAS>
__global__ __launch_bounds__(256) void k_gemm(const float* __restrict__ A,
                                              const _Float16* __restrict__ Bp,
                                              float* __restrict__ C,
                                              const float* __restrict__ bias,
                                              int M) {
    constexpr int NO  = NT * 16;
    constexpr int KCF = K / 32;           // full chunks
    const int lane    = threadIdx.x & 31;
    const int wave    = threadIdx.x >> 5;
    const int rowBase = blockIdx.y * 128 + wave * 16;
    const int m       = rowBase + (lane & 15);
    const int mld     = (m < M) ? m : (M - 1);          // clamp: loads always safe
    const float* __restrict__ Arow = A + (size_t)mld * K;
    const int kg = (lane >> 4) * 8;
    const _Float16* __restrict__ bbase =
        Bp + ((((size_t)blockIdx.x) * 32 + lane) << 4);
    constexpr size_t BSTRIDE = (size_t)NT * 32 * 16;    // halves per k-chunk

    v8f acc = {};
#pragma unroll
    for (int kc = 0; kc < KCF; ++kc) {
        const int kb = kc * 32 + kg;
        v16h a;
        cvt8(a, 0, Arow + kb);
        cvt8(a, 8, Arow + kb + 16);
        v16h b = *(const v16h*)(bbase + (size_t)kc * BSTRIDE);
        acc = __builtin_amdgcn_wmma_f32_16x16x32_f16(
                  false, a, false, b, (short)0, acc, false, false);
    }
    if constexpr ((K % 32) != 0) {        // branch-free tail: clamp + select
        const int kb = KCF * 32 + kg;
        v16h a;
#pragma unroll
        for (int j = 0; j < 8; ++j) {
            int k0 = kb + j;
            int k1 = kb + 16 + j;
            float v0 = Arow[k0 < K ? k0 : K - 1];
            float v1 = Arow[k1 < K ? k1 : K - 1];
            a[j]     = (k0 < K) ? (_Float16)v0 : (_Float16)0.0f;
            a[8 + j] = (k1 < K) ? (_Float16)v1 : (_Float16)0.0f;
        }
        v16h b = *(const v16h*)(bbase + (size_t)KCF * BSTRIDE);
        acc = __builtin_amdgcn_wmma_f32_16x16x32_f16(
                  false, a, false, b, (short)0, acc, false, false);
    }

    const int n  = blockIdx.x * 16 + (lane & 15);
    const int mb = rowBase + (lane >> 4) * 8;
    float bv = 0.0f;
    if constexpr (HASBIAS) bv = bias[n];
#pragma unroll
    for (int r = 0; r < 8; ++r) {
        int mm = mb + r;
        if (mm < M) {
            float v = acc[r] + bv;
            if constexpr (RELU) v = fmaxf(v, 0.0f);
            C[(size_t)mm * NO + n] = v;
        }
    }
}

// fc1 over virtual concat [x | h1 | h2] (K = 200+128+128 = 456).
// All region boundaries (200, 328, 456) are 8-aligned -> each 8-wide K group
// maps to exactly one source.  With the k-chunk loop fully unrolled, the region
// choice folds to a constant for 12/15 chunks; straddling chunks use a
// cndmask pointer select (no branch).
__device__ __forceinline__ void catsel(v16h& d, int off,
                                       const float* __restrict__ xr,
                                       const float* __restrict__ h1r,
                                       const float* __restrict__ h2r, int k0) {
    const float* p = (k0 < NFEAT)          ? (xr  + k0)
                   : (k0 < NFEAT + NHID)   ? (h1r + (k0 - NFEAT))
                                           : (h2r + (k0 - NFEAT - NHID));
    cvt8(d, off, p);
}

__global__ __launch_bounds__(256) void k_gemm_cat(const float* __restrict__ x,
                                                  const float* __restrict__ h1,
                                                  const float* __restrict__ h2,
                                                  const _Float16* __restrict__ Bp,
                                                  float* __restrict__ C,
                                                  const float* __restrict__ bias,
                                                  int M) {
    constexpr int K   = NFEAT + 2 * NHID;  // 456
    constexpr int KCF = K / 32;            // 14 full chunks
    constexpr int B1  = NFEAT;             // 200
    constexpr int B2  = NFEAT + NHID;      // 328
    const int lane    = threadIdx.x & 31;
    const int wave    = threadIdx.x >> 5;
    const int rowBase = blockIdx.y * 128 + wave * 16;
    const int m       = rowBase + (lane & 15);
    const int mld     = (m < M) ? m : (M - 1);
    const float* __restrict__ xrow  = x  + (size_t)mld * NFEAT;
    const float* __restrict__ h1row = h1 + (size_t)mld * NHID;
    const float* __restrict__ h2row = h2 + (size_t)mld * NHID;
    const int kg = (lane >> 4) * 8;
    const _Float16* __restrict__ bbase =
        Bp + ((((size_t)blockIdx.x) * 32 + lane) << 4);
    constexpr size_t BSTRIDE = (size_t)8 * 32 * 16;

    v8f acc = {};
#pragma unroll
    for (int kc = 0; kc < KCF; ++kc) {
        const int kb = kc * 32 + kg;
        v16h a;
        if (kc * 32 + 31 < B1) {                       // pure x chunk
            cvt8(a, 0, xrow + kb);
            cvt8(a, 8, xrow + kb + 16);
        } else if (kc * 32 >= B1 && kc * 32 + 31 < B2) { // pure h1 chunk
            cvt8(a, 0, h1row + kb - B1);
            cvt8(a, 8, h1row + kb - B1 + 16);
        } else if (kc * 32 >= B2) {                    // pure h2 chunk
            cvt8(a, 0, h2row + kb - B2);
            cvt8(a, 8, h2row + kb - B2 + 16);
        } else {                                       // straddles a boundary
            catsel(a, 0, xrow, h1row, h2row, kb);
            catsel(a, 8, xrow, h1row, h2row, kb + 16);
        }
        v16h b = *(const v16h*)(bbase + (size_t)kc * BSTRIDE);
        acc = __builtin_amdgcn_wmma_f32_16x16x32_f16(
                  false, a, false, b, (short)0, acc, false, false);
    }
    {   // tail chunk kc=14 covers k 448..479; only group0 with kg=0 is valid
        const int kb = KCF * 32 + kg;
        v16h a;
#pragma unroll
        for (int j = 0; j < 8; ++j) {
            int k  = kb + j;
            int kk = (k < K) ? (k - B2) : (NHID - 8);  // clamp inside h2 row
            float v = h2row[kk];
            a[j]     = (k < K) ? (_Float16)v : (_Float16)0.0f;
            a[8 + j] = (_Float16)0.0f;                 // kb+16 >= 464 always OOB
        }
        v16h b = *(const v16h*)(bbase + (size_t)KCF * BSTRIDE);
        acc = __builtin_amdgcn_wmma_f32_16x16x32_f16(
                  false, a, false, b, (short)0, acc, false, false);
    }

    const int n  = blockIdx.x * 16 + (lane & 15);
    const int mb = rowBase + (lane >> 4) * 8;
    const float bv = bias[n];
#pragma unroll
    for (int r = 0; r < 8; ++r) {
        int mm = mb + r;
        if (mm < M)
            C[(size_t)mm * NHID + n] = fmaxf(acc[r] + bv, 0.0f);
    }
}

// ---------------------------------------------------------------- aggregation
// one wave per edge; each lane moves 4 of the 128 features (float4 + 4 atomics)
__global__ __launch_bounds__(256) void k_scatter(const int* __restrict__ src,
                                                 const int* __restrict__ dst,
                                                 const float* __restrict__ w,
                                                 const float* __restrict__ dinv,
                                                 const float* __restrict__ h,
                                                 float* __restrict__ agg, int e) {
    int gw   = (blockIdx.x * 256 + threadIdx.x) >> 5;   // edge id
    int lane = threadIdx.x & 31;
    if (gw >= e) return;
    int s = src[gw], d = dst[gw];
    float nrm = dinv[s] * w[gw] * dinv[d];
    const float4 v = *(const float4*)(h + (size_t)s * NHID + lane * 4);
    float* o = agg + (size_t)d * NHID + lane * 4;
    atomicAdd(o + 0, v.x * nrm);
    atomicAdd(o + 1, v.y * nrm);
    atomicAdd(o + 2, v.z * nrm);
    atomicAdd(o + 3, v.w * nrm);
}

// agg[i][f] = relu(agg[i][f] + h[i][f]*dinv[i]^2 + b[f])   (self loop + bias + relu)
__global__ __launch_bounds__(256) void k_epilogue(float* __restrict__ agg,
                                                  const float* __restrict__ h,
                                                  const float* __restrict__ dinv,
                                                  const float* __restrict__ b, int n) {
    int t = blockIdx.x * 256 + threadIdx.x;
    if (t >= n * NHID) return;
    int i = t >> 7, f = t & 127;
    float di = dinv[i];
    agg[t] = fmaxf(agg[t] + h[t] * di * di + b[f], 0.0f);
}

// ---------------------------------------------------------------- batch norm
#define STAT_CHUNK 512
__global__ __launch_bounds__(256) void k_stats(const float* __restrict__ h,
                                               float* __restrict__ stats, int n) {
    int c    = threadIdx.x & 127;
    int half = threadIdx.x >> 7;
    int rEnd = (blockIdx.x + 1) * STAT_CHUNK; if (rEnd > n) rEnd = n;
    float s = 0.f, q = 0.f;
    for (int r = blockIdx.x * STAT_CHUNK + half; r < rEnd; r += 2) {
        float v = h[(size_t)r * NHID + c];
        s += v; q += v * v;
    }
    __shared__ float sh[512];
    sh[threadIdx.x] = s; sh[256 + threadIdx.x] = q;
    __syncthreads();
    if (half == 0) {
        s = sh[threadIdx.x] + sh[threadIdx.x + 128];
        q = sh[256 + threadIdx.x] + sh[256 + threadIdx.x + 128];
        atomicAdd(&stats[c], s);
        atomicAdd(&stats[128 + c], q);
    }
}

__global__ __launch_bounds__(256) void k_bn(float* __restrict__ h,
                                            const float* __restrict__ stats,
                                            const float* __restrict__ g,
                                            const float* __restrict__ beta, int n) {
    int t = blockIdx.x * 256 + threadIdx.x;
    if (t >= n * NHID) return;
    int c = t & 127;
    float invn = 1.0f / (float)n;
    float mean = stats[c] * invn;
    float var  = stats[128 + c] * invn - mean * mean;
    h[t] = g[c] * (h[t] - mean) * rsqrtf(var + BN_EPS) + beta[c];
}

// ---------------------------------------------------------------- launcher

extern "C" void kernel_launch(void* const* d_in, const int* in_sizes, int n_in,
                              void* d_out, int out_size, void* d_ws, size_t ws_size,
                              hipStream_t stream) {
    const float* x      = (const float*)d_in[0];
    const int*   adj    = (const int*)  d_in[1];   // [2][E]; row0=src, row1=dst
    const float* weight = (const float*)d_in[2];
    const float* W1     = (const float*)d_in[3];
    const float* b1     = (const float*)d_in[4];
    const float* W2     = (const float*)d_in[5];
    const float* b2     = (const float*)d_in[6];
    const float* g1     = (const float*)d_in[7];
    const float* beta1  = (const float*)d_in[8];
    const float* g2     = (const float*)d_in[9];
    const float* beta2  = (const float*)d_in[10];
    const float* fc1W   = (const float*)d_in[11];
    const float* fc1b   = (const float*)d_in[12];
    const float* fc2W   = (const float*)d_in[13];
    const float* fc2b   = (const float*)d_in[14];
    float* out = (float*)d_out;

    const int* src = adj;
    const int* dst = adj + GE;

    // ---- workspace carve-up (256B aligned)
    char* cur = (char*)d_ws;
    auto alloc = [&](size_t bytes) {
        void* p = (void*)cur;
        cur += (bytes + 255) & ~(size_t)255;
        return p;
    };
    float*    dinv  = (float*)alloc((size_t)GN * 4);
    float*    stats = (float*)alloc(256 * 4);
    // packed B sizes: kchunks * (Nout/16) * 32 * 16 halves
    _Float16* w1p  = (_Float16*)alloc((size_t)7  * 8 * 512 * 2);   // K=200
    _Float16* w2p  = (_Float16*)alloc((size_t)4  * 8 * 512 * 2);   // K=128
    _Float16* fc1p = (_Float16*)alloc((size_t)15 * 8 * 512 * 2);   // K=456
    _Float16* fc2p = (_Float16*)alloc((size_t)4  * 4 * 512 * 2);   // K=128, Nout=64
    float* tmp = (float*)alloc((size_t)GN * NHID * 4);   // pre-agg h / fc1 out
    float* h1  = (float*)alloc((size_t)GN * NHID * 4);
    float* h2  = (float*)alloc((size_t)GN * NHID * 4);
    (void)ws_size; (void)n_in; (void)in_sizes; (void)out_size;

    const int gN       = (GN + 255) / 256;              // 391
    const int gEdge    = (GE + 255) / 256;              // 6250
    const int gNH      = (GN * NHID) / 256;             // 50000
    const int gNH4     = (GN * NHID / 4 + 255) / 256;   // 12500
    const int gScatter = GE / 8;                        // 200000 (wave per edge)
    const int gStats   = (GN + STAT_CHUNK - 1) / STAT_CHUNK;  // 196
    const dim3 gemmGrid((unsigned)(NHID / 16), (unsigned)((GN + 127) / 128));  // (8, 782)
    const dim3 fc2Grid((unsigned)(NOUT / 16),  (unsigned)((GN + 127) / 128));  // (4, 782)

    // ---- dinv = rsqrt(1 + weighted in-degree)
    k_fill <<<gN, 256, 0, stream>>>(dinv, 1.0f, GN);
    k_deg  <<<gEdge, 256, 0, stream>>>(dst, weight, dinv, GE);
    k_rsqrt<<<gN, 256, 0, stream>>>(dinv, GN);

    // ---- pack all weights into WMMA B-fragment layout
    k_packB<<<(7*8*512  + 255)/256, 256, 0, stream>>>(W1,   w1p,  NFEAT,          NHID);
    k_packB<<<(4*8*512  + 255)/256, 256, 0, stream>>>(W2,   w2p,  NHID,           NHID);
    k_packB<<<(15*8*512 + 255)/256, 256, 0, stream>>>(fc1W, fc1p, NFEAT + 2*NHID, NHID);
    k_packB<<<(4*4*512  + 255)/256, 256, 0, stream>>>(fc2W, fc2p, NHID,           NOUT);

    // ---- layer 1: h = x@W1 ; aggregate ; +b1 ; relu ; BN
    k_gemm<NFEAT, 8, false, false><<<gemmGrid, 256, 0, stream>>>(x, w1p, tmp, nullptr, GN);
    k_zero4<<<gNH4, 256, 0, stream>>>((float4*)h1, GN * NHID / 4);
    k_scatter<<<gScatter, 256, 0, stream>>>(src, dst, weight, dinv, tmp, h1, GE);
    k_epilogue<<<gNH, 256, 0, stream>>>(h1, tmp, dinv, b1, GN);
    k_zero4<<<1, 256, 0, stream>>>((float4*)stats, 64);
    k_stats<<<gStats, 256, 0, stream>>>(h1, stats, GN);
    k_bn   <<<gNH, 256, 0, stream>>>(h1, stats, g1, beta1, GN);

    // ---- layer 2
    k_gemm<NHID, 8, false, false><<<gemmGrid, 256, 0, stream>>>(h1, w2p, tmp, nullptr, GN);
    k_zero4<<<gNH4, 256, 0, stream>>>((float4*)h2, GN * NHID / 4);
    k_scatter<<<gScatter, 256, 0, stream>>>(src, dst, weight, dinv, tmp, h2, GE);
    k_epilogue<<<gNH, 256, 0, stream>>>(h2, tmp, dinv, b2, GN);
    k_zero4<<<1, 256, 0, stream>>>((float4*)stats, 64);
    k_stats<<<gStats, 256, 0, stream>>>(h2, stats, GN);
    k_bn   <<<gNH, 256, 0, stream>>>(h2, stats, g2, beta2, GN);

    // ---- MLP head: relu([x|h1|h2]@fc1 + b) -> relu(@fc2 + b)
    k_gemm_cat<<<gemmGrid, 256, 0, stream>>>(x, h1, h2, fc1p, tmp, fc1b, GN);
    k_gemm<NHID, 4, true, true><<<fc2Grid, 256, 0, stream>>>(tmp, fc2p, out, fc2b, GN);
}
